// UpSample_77996606095599
// MI455X (gfx1250) — matrix operand order repaired and checked by reading, not compiled, not run on hardware
//
#include <hip/hip_runtime.h>
#include <hip/hip_bf16.h>
#include <stdint.h>

#define NIN   16384
#define NOUT  131072
#define CCH   64

typedef _Float16 v8h  __attribute__((ext_vector_type(8)));
typedef _Float16 v16h __attribute__((ext_vector_type(16)));
typedef float    v8f  __attribute__((ext_vector_type(8)));

#if defined(__has_builtin)
#if __has_builtin(__builtin_amdgcn_global_load_async_to_lds_b128)
#define HAVE_ASYNC_LDS 1
#endif
#if __has_builtin(__builtin_amdgcn_s_wait_asynccnt)
#define HAVE_WAIT_ASYNC 1
#endif
#endif

#if defined(HAVE_ASYNC_LDS)
typedef int v4i_ __attribute__((vector_size(16)));
using gv4 = __attribute__((address_space(1))) v4i_;   // global (as1) 16B vector
using lv4 = __attribute__((address_space(3))) v4i_;   // LDS (as3) 16B vector
#endif

__device__ __forceinline__ unsigned fkey(float f) {
    unsigned u = __float_as_uint(f);
    return u ^ ((u & 0x80000000u) ? 0xFFFFFFFFu : 0x80000000u); // order-preserving key
}

// Stage `halfs` f16 elements from global to LDS; 16B chunks spread over the block.
__device__ __forceinline__ void stage_frag(const _Float16* __restrict__ g,
                                           _Float16* l, int halfs) {
    const int chunks = halfs >> 3; // 16B chunks
    for (int c = threadIdx.x; c < chunks; c += blockDim.x) {
#if defined(HAVE_ASYNC_LDS)
        __builtin_amdgcn_global_load_async_to_lds_b128(
            (gv4*)(uintptr_t)(g + c * 8),
            (lv4*)(unsigned)(uintptr_t)(l + c * 8),
            0, 0);
#else
        *(v8h*)(l + c * 8) = *(const v8h*)(g + c * 8);
#endif
    }
}

__device__ __forceinline__ void async_join() {
#if defined(HAVE_ASYNC_LDS)
#if defined(HAVE_WAIT_ASYNC)
    __builtin_amdgcn_s_wait_asynccnt(0);
#else
    asm volatile("s_wait_asynccnt 0x0" ::: "memory");
#endif
#endif
    __syncthreads();
}

// ---------------- elementwise helpers ----------------
__global__ void zero_f16(_Float16* p, int n) {
    int i = blockIdx.x * blockDim.x + threadIdx.x;
    if (i < n) p[i] = (_Float16)0.f;
}
__global__ void cvt_f32_to_f16(const float* __restrict__ s, _Float16* __restrict__ d, int n) {
    int i = blockIdx.x * blockDim.x + threadIdx.x;
    if (i < n) d[i] = (_Float16)s[i];
}

// ---------------- weight repack into WMMA B-fragment order ----------------
// frag element idx = (((tap*KC + kc)*NC + nc) << 9) | (lane << 4) | e
// lane n holds column (nc*16 + n&15), K = kc*32 + (e>=8 ? 16:0) + (lane>=16 ? 8:0) + (e&7)
__global__ void repack_w(const float* __restrict__ W, _Float16* __restrict__ dst,
                         int taps, int Cin, int Cout, int KC, int NC) {
    int idx = blockIdx.x * blockDim.x + threadIdx.x;
    int total = taps * KC * NC * 512;
    if (idx >= total) return;
    int e    = idx & 15;
    int lane = (idx >> 4) & 31;
    int rest = idx >> 9;
    int nc   = rest % NC;  rest /= NC;
    int kcI  = rest % KC;
    int tap  = rest / KC;
    int K    = kcI * 32 + ((e >> 3) << 4) + ((lane >> 4) << 3) + (e & 7);
    int col  = nc * 16 + (lane & 15);
    float v  = (K < Cin && col < Cout) ? W[((size_t)tap * Cin + K) * Cout + col] : 0.f;
    dst[idx] = (_Float16)v;
}

// ---------------- WMMA sparse-conv / GEMM kernel ----------------
// One wave -> one 16-row M tile, NC 16-col N tiles, accumulating over taps x KC k-chunks.
// Weight fragments for each tap are async-staged into LDS (double-buffered) once per
// block and consumed by all 8 waves via ds_load. A rows are gathered per lane.
// UP=1: generative transpose conv: A rows are input rows, tap = blockIdx.y, scatter row*8+tap.
template<int KC, int NC, int UP, int RELU, int RESID, int PADZ>
__global__ __launch_bounds__(256)
void spconv(const _Float16* __restrict__ feats, int featStride,
            const int* __restrict__ nbr, int taps, int nrowsOut,
            const _Float16* __restrict__ wfrag,
            const float* __restrict__ bias,
            const _Float16* __restrict__ resid,
            _Float16* __restrict__ outp, int outStride, int colOff) {
    constexpr int FR = KC * NC * 512;          // halfs per tap fragment set
    __shared__ __align__(32) _Float16 wlds[2][FR];

    const int lane  = threadIdx.x & 31;
    const int wave  = threadIdx.x >> 5;
    const int mtile = blockIdx.x * 8 + wave;
    const int m0    = mtile * 16;
    const int lh    = lane & 15;
    const int q     = (lane >> 4) << 3;   // 0 or 8 (A fragment half-select)
    if (UP) wfrag += (size_t)blockIdx.y * FR;

    const int ntaps = UP ? 1 : taps;

    // prologue: stage tap 0
    stage_frag(wfrag, wlds[0], FR);
    async_join();

    v8f acc[NC] = {};
    for (int tap = 0; tap < ntaps; ++tap) {
        if (tap + 1 < ntaps)   // prefetch next tap's fragments into the other buffer
            stage_frag(wfrag + (size_t)(tap + 1) * FR, wlds[(tap + 1) & 1], FR);

        const _Float16* wl = wlds[tap & 1];

        int r;
        if (UP)          r = m0 + lh;
        else if (nbr)    r = nbr[(size_t)tap * nrowsOut + m0 + lh];  // may be NOUT -> zero sentinel row
        else             r = m0 + lh;
        const _Float16* ap = feats + (size_t)r * featStride;
#pragma unroll
        for (int kcI = 0; kcI < KC; ++kcI) {
            union { v16h v; struct { v8h lo, hi; } s; } A;
            A.s.lo = *(const v8h*)(ap + kcI * 32 + q);        // K = q..q+7
            A.s.hi = *(const v8h*)(ap + kcI * 32 + 16 + q);   // K = 16+q..16+q+7
#pragma unroll
            for (int nc = 0; nc < NC; ++nc) {
                v16h B = *(const v16h*)(wl + ((size_t)kcI * NC + nc) * 512 + lane * 16);
                acc[nc] = __builtin_amdgcn_wmma_f32_16x16x32_f16(
                    false, A.v, false, B, (short)0, acc[nc], false, false);
            }
        }
        // wait for the prefetch + make sure every wave finished reading before the
        // buffer is overwritten two taps from now
        async_join();
    }

    // epilogue: D layout: VGPR v -> M = v + (lane<16?0:8), N = lane&15
    const int vrow = (lane >> 4) << 3;
#pragma unroll
    for (int v = 0; v < 8; ++v) {
        int  mrow   = m0 + vrow + v;
        long rowOut = UP ? ((long)mrow * 8 + (long)blockIdx.y) : (long)mrow;
#pragma unroll
        for (int nc = 0; nc < NC; ++nc) {
            float x = acc[nc][v] + bias[nc * 16 + lh];
            if (RELU)  x = x > 0.f ? x : 0.f;
            if (RESID) x += (float)resid[rowOut * 64 + colOff + nc * 16 + lh];
            outp[rowOut * (long)outStride + colOff + nc * 16 + lh] = (_Float16)x;
        }
        if (PADZ) outp[rowOut * (long)outStride + colOff + 16 + lh] = (_Float16)0.f; // zero K-pad cols
    }
}

// ---------------- classifier conv (Cout=1) ----------------
__global__ void cls_kernel(const _Float16* __restrict__ feats, const int* __restrict__ nbr,
                           const float* __restrict__ Wcls, const float* __restrict__ bcls,
                           float* __restrict__ out_cls, int n) {
    int r = blockIdx.x * blockDim.x + threadIdx.x;
    if (r >= n) return;
    float acc = bcls[0];
    for (int t = 0; t < 27; ++t) {
        int s = nbr[(size_t)t * n + r];           // sentinel row n is zeroed
        const _Float16* fp = feats + (size_t)s * 64;
        const float* wp = Wcls + t * 64;
#pragma unroll
        for (int c = 0; c < 64; ++c) acc += (float)fp[c] * wp[c];
    }
    out_cls[r] = acc;
}

// ---------------- deterministic radix top-k select ----------------
// state: st[0]=prefix  st[1]=remaining  st[2]=threshold key ; st[8..263]=hist
__global__ void sel_init(unsigned* st, const int* numsP) {
    int i = threadIdx.x;
    if (i == 0) {
        int k = numsP[0];
        st[0] = 0u; st[1] = (unsigned)(k > 0 ? k : 1); st[2] = 0u;
    }
    st[8 + i] = 0u;
}
__global__ void sel_hist(const float* __restrict__ cls, unsigned* st, int n, int p) {
    int r = blockIdx.x * blockDim.x + threadIdx.x;
    if (r >= n) return;
    unsigned key = fkey(cls[r]);
    if (p > 0) { if ((key >> (32 - 8 * p)) != st[0]) return; }
    unsigned b = (key >> (24 - 8 * p)) & 255u;
    atomicAdd(&st[8 + b], 1u);
}
__global__ void sel_scan(unsigned* st, int p) {
    if (threadIdx.x || blockIdx.x) return;
    unsigned rem = st[1], cum = 0; int chosen = 0;
    for (int b = 255; b >= 0; --b) {
        unsigned c = st[8 + b];
        if (cum + c >= rem || b == 0) { chosen = b; rem -= cum; break; }
        cum += c;
    }
    st[0] = (st[0] << 8) | (unsigned)chosen;
    st[1] = rem;
    for (int b = 0; b < 256; ++b) st[8 + b] = 0u;
    if (p == 3) st[2] = st[0];
}

// ---------------- mask + prune writeback ----------------
__global__ void mask_prune(const _Float16* __restrict__ feats, const float* __restrict__ cls,
                           const unsigned char* __restrict__ gt, const unsigned* st,
                           float* __restrict__ pruned, float* __restrict__ maskO, int n) {
    int r = blockIdx.x * blockDim.x + threadIdx.x;
    if (r >= n) return;
    bool m = (fkey(cls[r]) >= st[2]) || (gt[r] != 0);
    maskO[r] = m ? 1.f : 0.f;
    const _Float16* fp = feats + (size_t)r * 64;
    float4* op = (float4*)(pruned + (size_t)r * 64);
#pragma unroll
    for (int c = 0; c < 64; c += 4) {
        float4 v;
        v.x = m ? (float)fp[c + 0] : 0.f;
        v.y = m ? (float)fp[c + 1] : 0.f;
        v.z = m ? (float)fp[c + 2] : 0.f;
        v.w = m ? (float)fp[c + 3] : 0.f;
        op[c >> 2] = v;
    }
}

extern "C" void kernel_launch(void* const* d_in, const int* in_sizes, int n_in,
                              void* d_out, int out_size, void* d_ws, size_t ws_size,
                              hipStream_t stream) {
    const float* x    = (const float*)d_in[0];
    const float* Wup  = (const float*)d_in[1];
    const float* bup  = (const float*)d_in[2];
    const float* Wc0  = (const float*)d_in[3];
    const float* bc0  = (const float*)d_in[4];
    const float* W00  = (const float*)d_in[5];
    const float* b00  = (const float*)d_in[6];
    const float* W01  = (const float*)d_in[7];
    const float* b01  = (const float*)d_in[8];
    const float* W10  = (const float*)d_in[9];
    const float* b10  = (const float*)d_in[10];
    const float* W11  = (const float*)d_in[11];
    const float* b11  = (const float*)d_in[12];
    const float* W12  = (const float*)d_in[13];
    const float* b12  = (const float*)d_in[14];
    const float* Wcls = (const float*)d_in[15];
    const float* bcls = (const float*)d_in[16];
    const int*   nbr  = (const int*)d_in[17];
    const unsigned char* gt = (const unsigned char*)d_in[18];
    const int*   numsP = (const int*)d_in[19];

    unsigned* st = (unsigned*)d_ws;                       // select state + hist (first 4 KB)
    _Float16* hb = (_Float16*)((char*)d_ws + 4096);
    size_t off = 0;
    _Float16* featA = hb + off; off += (size_t)(NOUT + 1) * 64;
    _Float16* featB = hb + off; off += (size_t)(NOUT + 1) * 64;
    _Float16* hbuf  = hb + off; off += (size_t)(NOUT + 1) * 32;
    _Float16* tbuf  = hb + off; off += (size_t)(NOUT + 1) * 32;
    _Float16* xh    = hb + off; off += (size_t)NIN * 64;
    _Float16* WupF  = hb + off; off += 8  * 2 * 4 * 512;
    _Float16* Wc0F  = hb + off; off += 27 * 2 * 4 * 512;
    _Float16* W00F  = hb + off; off += 3 * 27 * 2 * 1 * 512;
    _Float16* W01F  = hb + off; off += 3 * 27 * 1 * 2 * 512;
    _Float16* W10F  = hb + off; off += 3 * 1  * 2 * 1 * 512;
    _Float16* W11F  = hb + off; off += 3 * 27 * 1 * 1 * 512;
    _Float16* W12F  = hb + off; off += 3 * 1  * 1 * 2 * 512;

    // zero sentinel ("missing neighbor") rows
    zero_f16<<<1, 64, 0, stream>>>(featA + (size_t)NOUT * 64, 64);
    zero_f16<<<1, 64, 0, stream>>>(featB + (size_t)NOUT * 64, 64);
    zero_f16<<<1, 32, 0, stream>>>(hbuf  + (size_t)NOUT * 32, 32);
    zero_f16<<<1, 32, 0, stream>>>(tbuf  + (size_t)NOUT * 32, 32);

    cvt_f32_to_f16<<<(NIN * 64 + 255) / 256, 256, 0, stream>>>(x, xh, NIN * 64);

    auto rp = [&](const float* w, _Float16* dst, int taps, int Cin, int Cout, int KC, int NC) {
        int tot = taps * KC * NC * 512;
        repack_w<<<(tot + 255) / 256, 256, 0, stream>>>(w, dst, taps, Cin, Cout, KC, NC);
    };
    rp(Wup, WupF, 8, 64, 64, 2, 4);
    rp(Wc0, Wc0F, 27, 64, 64, 2, 4);
    for (int i = 0; i < 3; ++i) {
        rp(W00 + (size_t)i * 27 * 64 * 16, W00F + (size_t)i * 27648, 27, 64, 16, 2, 1);
        rp(W01 + (size_t)i * 27 * 16 * 32, W01F + (size_t)i * 27648, 27, 16, 32, 1, 2);
        rp(W10 + (size_t)i * 64 * 16,      W10F + (size_t)i * 1024,  1,  64, 16, 2, 1);
        rp(W11 + (size_t)i * 27 * 16 * 16, W11F + (size_t)i * 13824, 27, 16, 16, 1, 1);
        rp(W12 + (size_t)i * 16 * 32,      W12F + (size_t)i * 1024,  1,  16, 32, 1, 2);
    }

    // generative transpose conv (relu(up)) -> featA
    spconv<2, 4, 1, 1, 0, 0><<<dim3(NIN / 16 / 8, 8), 256, 0, stream>>>(
        xh, 64, nullptr, 1, NOUT, WupF, bup, nullptr, featA, 64, 0);
    // relu(sparse_conv Wc0) -> featB
    spconv<2, 4, 0, 1, 0, 0><<<dim3(NOUT / 16 / 8), 256, 0, stream>>>(
        featA, 64, nbr, 27, NOUT, Wc0F, bc0, nullptr, featB, 64, 0);

    _Float16* cur = featB;
    _Float16* alt = featA;
    for (int i = 0; i < 3; ++i) {
        // h = relu(conv27(cur, W00)) -> hbuf (16ch, padded to 32)
        spconv<2, 1, 0, 1, 0, 1><<<dim3(1024), 256, 0, stream>>>(
            cur, 64, nbr, 27, NOUT, W00F + (size_t)i * 27648, b00 + i * 16, nullptr, hbuf, 32, 0);
        // o0 = conv27(h, W01) + cur[:, :32] -> alt[:, :32]
        spconv<1, 2, 0, 0, 1, 0><<<dim3(1024), 256, 0, stream>>>(
            hbuf, 32, nbr, 27, NOUT, W01F + (size_t)i * 27648, b01 + i * 32, cur, alt, 64, 0);
        // t = relu(cur @ W10) -> tbuf (16ch padded)
        spconv<2, 1, 0, 1, 0, 1><<<dim3(1024), 256, 0, stream>>>(
            cur, 64, nullptr, 1, NOUT, W10F + (size_t)i * 1024, b10 + i * 16, nullptr, tbuf, 32, 0);
        // t = relu(conv27(t, W11)) -> hbuf (reuse; W01 already consumed it)
        spconv<1, 1, 0, 1, 0, 1><<<dim3(1024), 256, 0, stream>>>(
            tbuf, 32, nbr, 27, NOUT, W11F + (size_t)i * 13824, b11 + i * 16, nullptr, hbuf, 32, 0);
        // o1 = t @ W12 + cur[:, 32:] -> alt[:, 32:]
        spconv<1, 2, 0, 0, 1, 0><<<dim3(1024), 256, 0, stream>>>(
            hbuf, 32, nullptr, 1, NOUT, W12F + (size_t)i * 1024, b12 + i * 32, cur, alt, 64, 32);
        _Float16* t_ = cur; cur = alt; alt = t_;
    }

    float* out_cls = (float*)d_out;
    float* pruned  = (float*)d_out + NOUT;
    float* maskO   = (float*)d_out + NOUT + (size_t)NOUT * 64;

    cls_kernel<<<NOUT / 256, 256, 0, stream>>>(cur, nbr, Wcls, bcls, out_cls, NOUT);

    sel_init<<<1, 256, 0, stream>>>(st, numsP);
    for (int p = 0; p < 4; ++p) {
        sel_hist<<<NOUT / 256, 256, 0, stream>>>(out_cls, st, NOUT, p);
        sel_scan<<<1, 1, 0, stream>>>(st, p);
    }
    mask_prune<<<NOUT / 256, 256, 0, stream>>>(cur, out_cls, gt, st, pruned, maskO, NOUT);

    (void)in_sizes; (void)n_in; (void)out_size; (void)ws_size;
}
